// MaskedSelfAttention_29703993819976
// MI455X (gfx1250) — matrix operand order repaired and checked by reading, not compiled, not run on hardware
//
#include <hip/hip_runtime.h>
#include <hip/hip_bf16.h>

typedef __attribute__((ext_vector_type(16))) _Float16 v16h;
typedef __attribute__((ext_vector_type(8)))  _Float16 v8h;
typedef __attribute__((ext_vector_type(8)))  float    v8f;

#define EMBED    1024
#define HEADS    16
#define HEAD_DIM 64
#define NB       4
#define LSEQ     2048

// ---------------------------------------------------------------------------
// Fragment helpers (layouts per CDNA5 ISA §7.12.2, wave32)
// ---------------------------------------------------------------------------

static __device__ __forceinline__ v16h cat8(v8h lo, v8h hi) {
    return __builtin_shufflevector(lo, hi, 0,1,2,3,4,5,6,7,8,9,10,11,12,13,14,15);
}

// A fragment (16x32 f16): this lane holds row (lane%16); halves 0..15 are
// K = kbase + (lane/16)*16 .. +15 (contiguous)
static __device__ __forceinline__ v16h a_frag_f16(const _Float16* row, int kbase, int half) {
    v8h lo = *(const v8h*)(row + kbase + half * 16);
    v8h hi = *(const v8h*)(row + kbase + half * 16 + 8);
    return cat8(lo, hi);
}

// Same A fragment, but sourced from f32 data (converted on the fly)
static __device__ __forceinline__ v16h a_frag_f32(const float* row, int kbase, int half) {
    const float* p0 = row + kbase + half * 16;
    const float* p1 = row + kbase + half * 16 + 8;
    v16h r;
#pragma unroll
    for (int i = 0; i < 8; ++i) { r[i] = (_Float16)p0[i]; r[8 + i] = (_Float16)p1[i]; }
    return r;
}

static __device__ __forceinline__ v8f wmma_f16(v16h a, v16h b, v8f c) {
    // v_wmma_f32_16x16x32_f16
    return __builtin_amdgcn_wmma_f32_16x16x32_f16(false, a, false, b, (short)0, c, false, false);
}

// ---------------------------------------------------------------------------
// Tensor Data Mover (TDM): DMA global -> LDS, tracked by TENSORcnt.
// D# layout per CDNA5 ISA.
// ---------------------------------------------------------------------------
#if __has_builtin(__builtin_amdgcn_tensor_load_to_lds) && __has_builtin(__builtin_amdgcn_s_wait_tensorcnt)
#define USE_TDM 1
typedef unsigned int u32x4 __attribute__((ext_vector_type(4)));
typedef int          i32x4 __attribute__((ext_vector_type(4)));
typedef int          i32x8 __attribute__((ext_vector_type(8)));

// 2D (or 1D when tile_d1==0) tile load of 2-byte elements into LDS.
static __device__ __forceinline__ void tdm_load_to_lds(
        unsigned int lds_off, const void* gptr,
        unsigned int tile_d0, unsigned int tile_d1,
        unsigned long long stride0, unsigned int tensor_d0, unsigned int tensor_d1) {
    unsigned long long ga = (unsigned long long)(uintptr_t)gptr;
    u32x4 g0 = {0u, 0u, 0u, 0u};
    g0.x = 1u;                                               // count=1 (valid user D#)
    g0.y = lds_off;                                          // lds_addr (bytes)
    g0.z = (unsigned int)ga;                                 // global_addr[31:0]
    g0.w = (unsigned int)((ga >> 32) & 0x01FFFFFFu) | (2u << 30);  // addr[56:32], type=2
    i32x8 g1 = {0, 0, 0, 0, 0, 0, 0, 0};
    g1[0] = (int)(1u << 16);                                 // data_size=1 (2 bytes)
    g1[1] = (int)((tensor_d0 & 0xFFFFu) << 16);              // tensor_dim0[15:0]
    g1[2] = (int)((tensor_d0 >> 16) | ((tensor_d1 & 0xFFFFu) << 16));
    g1[3] = (int)((tensor_d1 >> 16) | ((tile_d0 & 0xFFFFu) << 16));
    g1[4] = (int)(tile_d1 & 0xFFFFu);                        // tile_dim1 (0=unused)
    g1[5] = (int)(unsigned int)(stride0 & 0xFFFFFFFFull);    // tensor_dim0_stride[31:0]
    g1[6] = (int)(unsigned int)((stride0 >> 32) & 0xFFFFull);
    i32x4 gz = {0, 0, 0, 0};
#if __clang_major__ >= 23
    i32x8 gz8 = {0, 0, 0, 0, 0, 0, 0, 0};
    __builtin_amdgcn_tensor_load_to_lds(g0, g1, gz, gz, gz8, 0);
#else
    __builtin_amdgcn_tensor_load_to_lds(g0, g1, gz, gz, 0);
#endif
}
#endif

// ---------------------------------------------------------------------------
// Kernel 0: f32 -> f16 weight conversion
// ---------------------------------------------------------------------------
__global__ void cvt_f32_f16_kernel(const float* __restrict__ src, _Float16* __restrict__ dst, int n) {
    int i = blockIdx.x * blockDim.x + threadIdx.x;
    if (i < n) dst[i] = (_Float16)src[i];
}

// ---------------------------------------------------------------------------
// Kernel 1: per-head projection  y = x @ W^T   (x: [N*L*H, 64] f32, W: [64,64])
// One wave per 16-row tile (16 rows == the 16 heads of one token).
// Q/K output: [N,H,L,64] f16.  V output: [N,H,64,L] f16 (transposed).
// ---------------------------------------------------------------------------
__global__ void proj_kernel(const float* __restrict__ x, const _Float16* __restrict__ Wh,
                            _Float16* __restrict__ outQK, _Float16* __restrict__ outVT,
                            int isV) {
    const int lane = threadIdx.x & 31;
    const int half = lane >> 4, lm = lane & 15;
    const int w = blockIdx.x * (blockDim.x >> 5) + (threadIdx.x >> 5);
    const int rbase = w * 16;                      // 8192 tiles * 16 = 131072 rows

    const float* arow = x + (size_t)(rbase + lm) * HEAD_DIM;
    v16h a0 = a_frag_f32(arow, 0, half);
    v16h a1 = a_frag_f32(arow, 32, half);

    v8f acc[4] = {v8f{}, v8f{}, v8f{}, v8f{}};
#pragma unroll
    for (int eb = 0; eb < 4; ++eb) {
        const _Float16* bp = Wh + (size_t)(eb * 16 + lm) * HEAD_DIM + half * 16;
        v16h b0 = *(const v16h*)(bp);
        v16h b1 = *(const v16h*)(bp + 32);
        acc[eb] = wmma_f16(a0, b0, acc[eb]);
        acc[eb] = wmma_f16(a1, b1, acc[eb]);
    }

#pragma unroll
    for (int eb = 0; eb < 4; ++eb) {
#pragma unroll
        for (int j = 0; j < 8; ++j) {
            int r = rbase + j + 8 * half;          // token-row index
            int e = eb * 16 + lm;
            int n = r >> 15;                       // r / (L*16)
            int l = (r >> 4) & (LSEQ - 1);
            int h = r & 15;
            _Float16 val = (_Float16)acc[eb][j];
            if (!isV)
                outQK[((((size_t)n * HEADS + h) * LSEQ + l) * HEAD_DIM) + e] = val;
            else
                outVT[((((size_t)n * HEADS + h) * HEAD_DIM + e) * LSEQ) + l] = val;
        }
    }
}

// ---------------------------------------------------------------------------
// Kernel 2: flash attention. Block = 8 waves = 8 q-tiles of the SAME (n,h).
// Each 64-key chunk of K (row-major) and V^T (strided 2D tile) is staged into
// LDS once per block — by the Tensor Data Mover when available — then all 8
// waves consume it. Online softmax per 16x64 S tile; P bounced through
// wave-private LDS into A-fragment form; 8+8 WMMAs per chunk per wave.
// ---------------------------------------------------------------------------
__global__ void attn_kernel(const _Float16* __restrict__ Qp, const _Float16* __restrict__ Kp,
                            const _Float16* __restrict__ VpT, const int* __restrict__ mask,
                            _Float16* __restrict__ attnOut) {
    __shared__ __align__(16) _Float16 Kc[64 * HEAD_DIM];      // [key][d]    8 KB
    __shared__ __align__(16) _Float16 Vc[HEAD_DIM * 64];      // [d][key]    8 KB
    __shared__ __align__(16) _Float16 Plds[8][16 * 64];       // per-wave P 16 KB

    const int lane = threadIdx.x & 31;
    const int half = lane >> 4, lm = lane & 15;
    const int warp = threadIdx.x >> 5;
    const int bh   = blockIdx.x >> 4;              // (n,h) id: 0..63
    const int n    = bh >> 4;
    const int h    = bh & 15;
    const int qbase = ((blockIdx.x & 15) * 8 + warp) * 16;

    const _Float16* Qrow  = Qp  + (((size_t)n * HEADS + h) * LSEQ + qbase + lm) * HEAD_DIM;
    const _Float16* Kbase = Kp  + ((size_t)n * HEADS + h) * LSEQ * HEAD_DIM;
    const _Float16* Vbase = VpT + ((size_t)n * HEADS + h) * (size_t)HEAD_DIM * LSEQ;

    v16h aq0 = a_frag_f16(Qrow, 0, half);
    v16h aq1 = a_frag_f16(Qrow, 32, half);

    float m[8], lsum[8];
    v8f o[4] = {v8f{}, v8f{}, v8f{}, v8f{}};
#pragma unroll
    for (int j = 0; j < 8; ++j) { m[j] = -INFINITY; lsum[j] = 0.f; }

    const float scale = 0.03125f;                  // 1/sqrt(EMBED) = 1/32
    _Float16* P = Plds[warp];

    for (int kb = 0; kb < LSEQ; kb += 64) {
        // ---- stage K/V chunk into LDS ----
#if defined(USE_TDM)
        if (warp == 0) {
            // K: 1D contiguous 64*64 halves
            tdm_load_to_lds((unsigned int)(uintptr_t)Kc, Kbase + (size_t)kb * HEAD_DIM,
                            64u * HEAD_DIM, 0u, 1ull << 20, 1u << 20, 0u);
            // V: 2D tile, 64 keys x 64 d-rows, row stride = LSEQ elements
            tdm_load_to_lds((unsigned int)(uintptr_t)Vc, Vbase + kb,
                            64u, (unsigned int)HEAD_DIM, (unsigned long long)LSEQ,
                            (unsigned int)LSEQ, 1u << 20);
            __builtin_amdgcn_s_wait_tensorcnt(0);
        }
#else
        {
            const uint4* gk = (const uint4*)(Kbase + (size_t)kb * HEAD_DIM);
            uint4* sk = (uint4*)Kc;
            for (int i = threadIdx.x; i < 512; i += 256) sk[i] = gk[i];
            for (int i = threadIdx.x; i < 512; i += 256) {
                int row = i >> 3, col = (i & 7) * 8;
                *(uint4*)&Vc[row * 64 + col] = *(const uint4*)(Vbase + (size_t)row * LSEQ + kb + col);
            }
        }
#endif
        __syncthreads();

        // ---- S = Q K^T for 64 keys: four 16x16 column tiles, K-dim = 64 ----
        v8f s[4];
#pragma unroll
        for (int c = 0; c < 4; ++c) {
            const _Float16* kr = Kc + (size_t)(c * 16 + lm) * HEAD_DIM + half * 16;
            v16h b0 = *(const v16h*)(kr);          // d 0..31
            v16h b1 = *(const v16h*)(kr + 32);     // d 32..63
            v8f sc = v8f{};
            sc = wmma_f16(aq0, b0, sc);
            sc = wmma_f16(aq1, b1, sc);
            s[c] = sc;
        }

        // ---- mask + scale (mask stored [K,Q]; element (q,k) = mask[k*L+q]) ----
        float sv[4][8];
#pragma unroll
        for (int c = 0; c < 4; ++c) {
            const int* mc = mask + (size_t)(kb + c * 16 + lm) * LSEQ + qbase + 8 * half;
#pragma unroll
            for (int j = 0; j < 8; ++j)
                sv[c][j] = (mc[j] == 0) ? -1e30f : s[c][j] * scale;
        }

        // ---- online softmax update ----
        float alpha[8];
#pragma unroll
        for (int j = 0; j < 8; ++j) {
            float rmax = fmaxf(fmaxf(sv[0][j], sv[1][j]), fmaxf(sv[2][j], sv[3][j]));
#pragma unroll
            for (int d = 1; d < 16; d <<= 1) rmax = fmaxf(rmax, __shfl_xor(rmax, d));
            float mn = fmaxf(m[j], rmax);
            int row = j + 8 * half;                // C/D layout row for this lane
            float rs = 0.f;
#pragma unroll
            for (int c = 0; c < 4; ++c) {
                float p = __expf(sv[c][j] - mn);
                rs += p;
                P[row * 64 + c * 16 + lm] = (_Float16)p;
            }
#pragma unroll
            for (int d = 1; d < 16; d <<= 1) rs += __shfl_xor(rs, d);
            alpha[j] = __expf(m[j] - mn);
            m[j] = mn;
            lsum[j] = lsum[j] * alpha[j] + rs;
        }

        // rescale running O
#pragma unroll
        for (int db = 0; db < 4; ++db)
#pragma unroll
            for (int j = 0; j < 8; ++j) o[db][j] *= alpha[j];

        // ---- O += P (16x64) x V (64x64): 2 key-groups x 4 d-blocks ----
#pragma unroll
        for (int kg = 0; kg < 2; ++kg) {
            v16h ap = a_frag_f16(P + lm * 64, kg * 32, half);
#pragma unroll
            for (int db = 0; db < 4; ++db) {
                const _Float16* vp = Vc + (size_t)(db * 16 + lm) * 64 + kg * 32 + half * 16;
                v16h bv = *(const v16h*)(vp);
                o[db] = wmma_f16(ap, bv, o[db]);
            }
        }
        __syncthreads();                           // all waves done before re-stage
    }

    // ---- normalize + store f16 [token, EMBED] for fc_out ----
#pragma unroll
    for (int db = 0; db < 4; ++db) {
#pragma unroll
        for (int j = 0; j < 8; ++j) {
            int q = qbase + j + 8 * half;
            int d = db * 16 + lm;
            float val = o[db][j] / lsum[j];
            attnOut[(size_t)(n * LSEQ + q) * EMBED + h * HEAD_DIM + d] = (_Float16)val;
        }
    }
}

// ---------------------------------------------------------------------------
// Kernel 3: fc_out  out = X @ Wo^T + bo   (X: [8192,1024] f16, Wo: [1024,1024])
// One wave per 64x16 output block (4 row-tiles share each B fragment).
// ---------------------------------------------------------------------------
__global__ void fcout_kernel(const _Float16* __restrict__ X, const _Float16* __restrict__ Woh,
                             const float* __restrict__ bias, float* __restrict__ out) {
    const int lane = threadIdx.x & 31;
    const int half = lane >> 4, lm = lane & 15;
    const int t = blockIdx.x * 8 + (threadIdx.x >> 5);   // 8192 waves
    const int ot = t & 63;                               // col tile (16 outs)
    const int rg = t >> 6;                               // row group (64 rows), 128 groups
    const int rbase = rg * 64, obase = ot * 16;

    const _Float16* arow = X + (size_t)(rbase + lm) * EMBED;
    const _Float16* brow = Woh + (size_t)(obase + lm) * EMBED;   // B[k=in][n=out]=Wo[out][in]

    v8f acc[4] = {v8f{}, v8f{}, v8f{}, v8f{}};
    for (int kc = 0; kc < EMBED; kc += 32) {
        v16h b = *(const v16h*)(brow + kc + half * 16);
#pragma unroll
        for (int rt = 0; rt < 4; ++rt) {
            v16h a = a_frag_f16(arow + (size_t)rt * 16 * EMBED, kc, half);
            acc[rt] = wmma_f16(a, b, acc[rt]);
        }
    }

    float bv = bias[obase + lm];
#pragma unroll
    for (int rt = 0; rt < 4; ++rt) {
#pragma unroll
        for (int j = 0; j < 8; ++j) {
            int r = rbase + rt * 16 + j + 8 * half;
            out[(size_t)r * EMBED + obase + lm] = acc[rt][j] + bv;
        }
    }
}

// ---------------------------------------------------------------------------
// Launch
// ---------------------------------------------------------------------------
extern "C" void kernel_launch(void* const* d_in, const int* in_sizes, int n_in,
                              void* d_out, int out_size, void* d_ws, size_t ws_size,
                              hipStream_t stream) {
    const float* values = (const float*)d_in[0];
    const float* keysIn = (const float*)d_in[1];
    const float* query  = (const float*)d_in[2];
    const int*   mask   = (const int*)d_in[3];
    const float* Wv     = (const float*)d_in[4];
    const float* Wk     = (const float*)d_in[5];
    const float* Wq     = (const float*)d_in[6];
    const float* Wo     = (const float*)d_in[7];
    const float* bo     = (const float*)d_in[8];
    float* out = (float*)d_out;

    // workspace layout (f16 tensors, all offsets well-aligned)
    char* ws = (char*)d_ws;
    const size_t SZ_HEAD = (size_t)NB * HEADS * LSEQ * HEAD_DIM * sizeof(_Float16); // 16 MB
    size_t off = 0;
    _Float16* Qp  = (_Float16*)(ws + off); off += SZ_HEAD;
    _Float16* Kp  = (_Float16*)(ws + off); off += SZ_HEAD;
    _Float16* VpT = (_Float16*)(ws + off); off += SZ_HEAD;
    _Float16* Xo  = (_Float16*)(ws + off); off += SZ_HEAD;      // attention output [token,1024]
    _Float16* Woh = (_Float16*)(ws + off); off += (size_t)EMBED * EMBED * sizeof(_Float16);
    _Float16* Whv = (_Float16*)(ws + off); off += (size_t)HEAD_DIM * HEAD_DIM * sizeof(_Float16);
    _Float16* Whk = (_Float16*)(ws + off); off += (size_t)HEAD_DIM * HEAD_DIM * sizeof(_Float16);
    _Float16* Whq = (_Float16*)(ws + off); off += (size_t)HEAD_DIM * HEAD_DIM * sizeof(_Float16);

    // 0) convert weights to f16
    cvt_f32_f16_kernel<<<(HEAD_DIM * HEAD_DIM + 255) / 256, 256, 0, stream>>>(Wv, Whv, HEAD_DIM * HEAD_DIM);
    cvt_f32_f16_kernel<<<(HEAD_DIM * HEAD_DIM + 255) / 256, 256, 0, stream>>>(Wk, Whk, HEAD_DIM * HEAD_DIM);
    cvt_f32_f16_kernel<<<(HEAD_DIM * HEAD_DIM + 255) / 256, 256, 0, stream>>>(Wq, Whq, HEAD_DIM * HEAD_DIM);
    cvt_f32_f16_kernel<<<(EMBED * EMBED + 255) / 256, 256, 0, stream>>>(Wo, Woh, EMBED * EMBED);

    // 1) projections: 131072 rows / 16 per wave / 8 waves per block = 1024 blocks
    proj_kernel<<<1024, 256, 0, stream>>>(query,  Whq, Qp, Qp,  0);
    proj_kernel<<<1024, 256, 0, stream>>>(keysIn, Whk, Kp, Kp,  0);
    proj_kernel<<<1024, 256, 0, stream>>>(values, Whv, VpT, VpT, 1);

    // 2) attention: 64 (n,h) x 16 blocks, each block = 8 q-tiles sharing K/V via LDS
    attn_kernel<<<1024, 256, 0, stream>>>(Qp, Kp, VpT, mask, Xo);

    // 3) fc_out: 128 row-groups x 64 col-tiles = 8192 waves -> 1024 blocks
    fcout_kernel<<<1024, 256, 0, stream>>>(Xo, Woh, bo, out);
}
